// MaskedMultiHeadSelfAttention_29411936043098
// MI455X (gfx1250) — matrix operand order repaired and checked by reading, not compiled, not run on hardware
//
#include <hip/hip_runtime.h>

typedef __attribute__((ext_vector_type(16))) _Float16 v16h;
typedef __attribute__((ext_vector_type(8)))  _Float16 v8h;
typedef __attribute__((ext_vector_type(8)))  float    v8f;

#define N_HEAD 12

__device__ __forceinline__ v8f wmma_f16(v16h a, v16h b, v8f c) {
  // D = A(16x32 f16) * B(32x16 f16) + C(16x16 f32)
  return __builtin_amdgcn_wmma_f32_16x16x32_f16(
      /*neg_a=*/false, a, /*neg_b=*/false, b,
      /*c_mod=*/(short)0, c, /*reuse_a=*/false, /*reuse_b=*/false);
}

// A-fragment (16x32, row-major source): lane = (m = lane&15, khalf = lane>>4)
// a[0..7]  = A[m][kb + khalf*16 + i] ; a[8..15] = A[m][kb + khalf*16 + 8 + i]
__device__ __forceinline__ v16h load_a_frag(const _Float16* row, int kb, int half) {
  v8h lo = *(const v8h*)(row + kb + half * 16);
  v8h hi = *(const v8h*)(row + kb + half * 16 + 8);
  v16h a;
#pragma unroll
  for (int i = 0; i < 8; ++i) { a[i] = lo[i]; a[i + 8] = hi[i]; }
  return a;
}

// B-fragment (32x16) from K-contiguous (transposed) storage:
// b[i] = colptr[kb + kgrp*16 + i]  -> one 32B vector load per lane
__device__ __forceinline__ v16h load_b_frag_contig(const _Float16* colptr, int kb, int half) {
  return *(const v16h*)(colptr + kb + half * 16);
}

__global__ void f32_to_f16_kernel(const float* __restrict__ src,
                                  _Float16* __restrict__ dst, size_t n) {
  size_t i = (size_t)blockIdx.x * blockDim.x + threadIdx.x;
  size_t stride = (size_t)gridDim.x * blockDim.x;
  for (; i < n; i += stride) dst[i] = (_Float16)src[i];
}

// LDS-tiled downconvert + transpose: src fp32 [K,N] -> dst f16 [N,K]
__global__ void transpose_f32_to_f16_kernel(const float* __restrict__ src,
                                            _Float16* __restrict__ dst,
                                            int K, int N) {
  __shared__ float tile[32][33];
  const int kb = blockIdx.y * 32, nb = blockIdx.x * 32;
  const int tx = threadIdx.x & 31;
  const int ty = threadIdx.x >> 5;        // 256 threads = 32x8
#pragma unroll
  for (int i = ty; i < 32; i += 8)
    tile[i][tx] = src[(size_t)(kb + i) * N + nb + tx];
  __syncthreads();
#pragma unroll
  for (int i = ty; i < 32; i += 8)
    dst[(size_t)(nb + i) * K + kb + tx] = (_Float16)tile[tx][i];
}

// qkv = x @ Wqkv + bqkv ; scatter q,k -> [B,H,T,D], v -> [B,H,D,T] (f16)
// Each wave: 16 rows x 64 cols (4 accumulators, A-fragment reused 4x).
// A: [B*T, C] f16 row-major; Bt: [3C, C] f16 (transposed weights)
__global__ void qkv_gemm_kernel(const _Float16* __restrict__ A,
                                const _Float16* __restrict__ Bt,
                                const float* __restrict__ bias,
                                _Float16* __restrict__ qh,
                                _Float16* __restrict__ kh,
                                _Float16* __restrict__ vh,
                                int T, int C, int H, int D) {
  const int K = C;
  const int lane = threadIdx.x & 31, wave = threadIdx.x >> 5;
  const int mt = blockIdx.y;
  const int ncol0 = (blockIdx.x * 4 + wave) * 64;   // 64-col strip per wave
  const int m = lane & 15, half = lane >> 4, n = lane & 15;

  const _Float16* arow = A + (size_t)(mt * 16 + m) * K;
  const _Float16* bcol[4];
#pragma unroll
  for (int j = 0; j < 4; ++j)
    bcol[j] = Bt + (size_t)(ncol0 + j * 16 + n) * K;

  v8f acc[4] = {};
  for (int kb = 0; kb < K; kb += 32) {
    __builtin_prefetch(bcol[0] + kb + 128, 0, 3);
    v16h a = load_a_frag(arow, kb, half);
#pragma unroll
    for (int j = 0; j < 4; ++j) {
      v16h b = load_b_frag_contig(bcol[j], kb, half);
      acc[j] = wmma_f16(a, b, acc[j]);
    }
  }

#pragma unroll
  for (int j = 0; j < 4; ++j) {
    const int ng = ncol0 + j * 16 + n;            // column in [0, 3C)
    const float bv = bias[ng];
    const int which = ng / C;                     // 0=q, 1=k, 2=v
    const int ch = ng % C;
    const int hh = ch / D, dd = ch % D;
#pragma unroll
    for (int r = 0; r < 8; ++r) {
      const int rowg = mt * 16 + r + 8 * half;    // global b*T + t
      const int bidx = rowg / T, t = rowg % T;
      const _Float16 val = (_Float16)(acc[j][r] + bv);
      if (which == 0)
        qh[(((size_t)bidx * H + hh) * T + t) * D + dd] = val;
      else if (which == 1)
        kh[(((size_t)bidx * H + hh) * T + t) * D + dd] = val;
      else  // V stored d-major so P@V B-fragments are contiguous
        vh[(((size_t)bidx * H + hh) * D + dd) * T + t] = val;
    }
  }
}

// Fused attention per (b*h, 16 query rows):
//  scores = (Q K^T)*scale -> 16xT fp32 strip in LDS, causal mask,
//  softmax (fp32 attn -> d_out, f16 copy -> second LDS region),
//  O = P V (f16 WMMA, LDS A-frags, contiguous global B-frags) -> oh [B,T,C].
// Dynamic LDS: 16*T*4 + 16*T*2 = 192 KB (fits CDNA5's 320 KB WGP LDS).
__global__ void attn_fused_kernel(const _Float16* __restrict__ qh,
                                  const _Float16* __restrict__ kh,
                                  const _Float16* __restrict__ vh,
                                  float* __restrict__ attn_out,
                                  _Float16* __restrict__ oh,
                                  int H, int T, int D) {
  extern __shared__ float s_scores[];                    // [16][T] fp32
  _Float16* s_pf16 = (_Float16*)(s_scores + 16 * T);     // [16][T] f16
  const int qt = blockIdx.x;
  const int bh = blockIdx.y;
  const int b = bh / H, h = bh % H;
  const int lane = threadIdx.x & 31, wave = threadIdx.x >> 5;
  const int m = lane & 15, half = lane >> 4, n = lane & 15;
  const int q0 = qt * 16;
  const float scale = 0.125f;                            // 1/sqrt(64)

  const _Float16* Q  = qh + (size_t)bh * T * D;          // [T,D]
  const _Float16* Km = kh + (size_t)bh * T * D;          // [T,D]
  const _Float16* Vt = vh + (size_t)bh * D * T;          // [D,T]

  // ---- Phase 1: 16x16 score tiles across keys (8 waves strided) ----
  // Q fragments are kt-invariant: hoist both k-steps out of the loop.
  const _Float16* qrow = Q + (size_t)(q0 + m) * D;
  const v16h aq0 = load_a_frag(qrow, 0, half);
  const v16h aq1 = load_a_frag(qrow, 32, half);
  for (int kt = wave; kt < T / 16; kt += 8) {
    const int kc0 = kt * 16;
    v8f c = {};
    if (kc0 <= q0 + 15) {                                // tile has unmasked entries
      const _Float16* krow = Km + (size_t)(kc0 + n) * D;
      c = wmma_f16(aq0, load_b_frag_contig(krow, 0, half), c);
      c = wmma_f16(aq1, load_b_frag_contig(krow, 32, half), c);
    }
#pragma unroll
    for (int r = 0; r < 8; ++r) {
      const int row = r + 8 * half;
      const int col = kc0 + n;
      float s = c[r] * scale;
      if (col > q0 + row) s = -__builtin_inff();
      s_scores[row * T + col] = s;
    }
  }
  __syncthreads();

  // ---- Phase 2: row softmax (2 rows/wave); write fp32 attn + f16 LDS copy ----
  for (int row = wave; row < 16; row += 8) {
    float* srow = s_scores + row * T;
    float mx = -__builtin_inff();
    for (int c0 = lane; c0 < T; c0 += 32) mx = fmaxf(mx, srow[c0]);
#pragma unroll
    for (int off = 16; off > 0; off >>= 1) mx = fmaxf(mx, __shfl_xor(mx, off, 32));
    float sum = 0.0f;
    for (int c0 = lane; c0 < T; c0 += 32) {
      float e = __expf(srow[c0] - mx);
      srow[c0] = e;
      sum += e;
    }
#pragma unroll
    for (int off = 16; off > 0; off >>= 1) sum += __shfl_xor(sum, off, 32);
    const float inv = 1.0f / sum;
    float* arow = attn_out + ((size_t)bh * T + (q0 + row)) * T;
    for (int c0 = lane; c0 < T; c0 += 32) {
      const float p = srow[c0] * inv;
      arow[c0] = p;                          // fp32 attention output
      s_pf16[row * T + c0] = (_Float16)p;    // f16 operand for P@V
    }
  }
  __syncthreads();

  // ---- Phase 3: O[16,64] = P[16,:kend] @ V[:kend,64]; waves 0..3 ----
  if (wave < 4) {
    const int d0 = wave * 16;
    const _Float16* prow = s_pf16 + m * T;               // LDS, contiguous
    const _Float16* vcol = Vt + (size_t)(d0 + n) * T;    // global, contiguous
    const int kend = ((q0 + 16 + 31) >> 5) << 5;         // causal frontier
    v8f c = {};
    for (int kb = 0; kb < kend; kb += 32) {
      v16h a = load_a_frag(prow, kb, half);              // ds_load_b128 x2
      v16h bfr = load_b_frag_contig(vcol, kb, half);     // global_load_b128 x2
      c = wmma_f16(a, bfr, c);
    }
    const int C = H * D;
#pragma unroll
    for (int r = 0; r < 8; ++r) {
      const int t = q0 + r + 8 * half;
      oh[((size_t)b * T + t) * C + h * D + d0 + n] = (_Float16)c[r];
    }
  }
}

// out = oh @ Wo + bo  (fp32 to d_out); Bt: [N,K] transposed weights.
// Each wave: 16 rows x 64 cols (A-fragment reused 4x).
__global__ void proj_gemm_kernel(const _Float16* __restrict__ A,
                                 const _Float16* __restrict__ Bt,
                                 const float* __restrict__ bias,
                                 float* __restrict__ out,
                                 int N, int K) {
  const int lane = threadIdx.x & 31, wave = threadIdx.x >> 5;
  const int mt = blockIdx.y;
  const int ncol0 = (blockIdx.x * 4 + wave) * 64;
  const int m = lane & 15, half = lane >> 4, n = lane & 15;

  const _Float16* arow = A + (size_t)(mt * 16 + m) * K;
  const _Float16* bcol[4];
#pragma unroll
  for (int j = 0; j < 4; ++j)
    bcol[j] = Bt + (size_t)(ncol0 + j * 16 + n) * K;

  v8f acc[4] = {};
  for (int kb = 0; kb < K; kb += 32) {
    __builtin_prefetch(bcol[0] + kb + 128, 0, 3);
    v16h a = load_a_frag(arow, kb, half);
#pragma unroll
    for (int j = 0; j < 4; ++j) {
      v16h b = load_b_frag_contig(bcol[j], kb, half);
      acc[j] = wmma_f16(a, b, acc[j]);
    }
  }

#pragma unroll
  for (int j = 0; j < 4; ++j) {
    const int ng = ncol0 + j * 16 + n;
    const float bv = bias[ng];
#pragma unroll
    for (int r = 0; r < 8; ++r) {
      const int rowg = mt * 16 + r + 8 * half;
      out[(size_t)rowg * N + ng] = acc[j][r] + bv;
    }
  }
}

extern "C" void kernel_launch(void* const* d_in, const int* in_sizes, int n_in,
                              void* d_out, int out_size, void* d_ws, size_t ws_size,
                              hipStream_t stream) {
  (void)in_sizes; (void)n_in; (void)out_size; (void)ws_size;
  const float* x    = (const float*)d_in[0];
  const float* Wqkv = (const float*)d_in[1];
  const float* bqkv = (const float*)d_in[2];
  const float* Wo   = (const float*)d_in[3];
  const float* bo   = (const float*)d_in[4];

  const int Bsz = 2, T = 2048, C = 768, H = N_HEAD, D = C / H;
  const size_t nX    = (size_t)Bsz * T * C;   // 3,145,728
  const size_t nWqkv = (size_t)C * 3 * C;     // 1,769,472
  const size_t nWo   = (size_t)C * C;         //   589,824

  _Float16* ws     = (_Float16*)d_ws;
  _Float16* xh     = ws;                      // [B*T, C]
  _Float16* wqkvt  = xh + nX;                 // [3C, C]  (transposed)
  _Float16* wot    = wqkvt + nWqkv;           // [C, C]   (transposed)
  _Float16* qh     = wot + nWo;               // [B,H,T,D]
  _Float16* kh     = qh + nX;                 // [B,H,T,D]
  _Float16* vh     = kh + nX;                 // [B,H,D,T] (d-major)
  _Float16* oh     = vh + nX;                 // [B,T,C]
  // total ~36 MB workspace

  float* out  = (float*)d_out;                // [B,T,C]
  float* attn = out + nX;                     // [B,H,T,T]

  f32_to_f16_kernel<<<2048, 256, 0, stream>>>(x, xh, nX);
  transpose_f32_to_f16_kernel<<<dim3(3 * C / 32, C / 32), 256, 0, stream>>>(
      Wqkv, wqkvt, C, 3 * C);
  transpose_f32_to_f16_kernel<<<dim3(C / 32, C / 32), 256, 0, stream>>>(
      Wo, wot, C, C);

  // M=B*T=4096 rows, N=3C=2304 cols: 256 cols/block -> grid (9, 256)
  qkv_gemm_kernel<<<dim3(3 * C / 256, Bsz * T / 16), 128, 0, stream>>>(
      xh, wqkvt, bqkv, qh, kh, vh, T, C, H, D);

  // one block per (16 query rows, b*h); 8 waves; 192 KB dynamic LDS
  attn_fused_kernel<<<dim3(T / 16, Bsz * H), 256,
                      (size_t)16 * T * (sizeof(float) + sizeof(_Float16)), stream>>>(
      qh, kh, vh, attn, oh, H, T, D);

  // out projection: M=4096, N=K=768, 256 cols/block -> grid (3, 256)
  proj_gemm_kernel<<<dim3(C / 256, Bsz * T / 16), 128, 0, stream>>>(
      oh, wot, bo, out, C, C);
}
